// ConCatMessage_80556406604248
// MI455X (gfx1250) — compile-verified, matrix operand order checked by reading, not run on hardware
//
#include <hip/hip_runtime.h>
#include <hip/hip_bf16.h>
#include <math.h>

typedef __attribute__((ext_vector_type(16))) _Float16 v16h;
typedef __attribute__((ext_vector_type(8)))  float    v8f;

#define B_EDGES   131072
#define N_USERS   500000
#define N_CAS     100000
#define D_MSG     128
#define IN_DIM    256

// d_out layout (reference tuple order, flattened):
static constexpr size_t OFF_SRC_M  = 0;
static constexpr size_t OFF_SRC_T  = OFF_SRC_M  + (size_t)N_USERS * D_MSG;
static constexpr size_t OFF_SRC_MK = OFF_SRC_T  + (size_t)N_USERS;
static constexpr size_t OFF_DST_M  = OFF_SRC_MK + (size_t)N_USERS;
static constexpr size_t OFF_DST_T  = OFF_DST_M  + (size_t)N_USERS * D_MSG;
static constexpr size_t OFF_DST_MK = OFF_DST_T  + (size_t)N_USERS;
static constexpr size_t OFF_CAS_M  = OFF_DST_MK + (size_t)N_USERS;
static constexpr size_t OFF_CAS_T  = OFF_CAS_M  + (size_t)N_CAS * D_MSG;
static constexpr size_t OFF_CAS_MK = OFF_CAS_T  + (size_t)N_CAS;

// Workspace layout (all regions 16B-aligned):
//   [0, 192KB)      packed f16 weights
//   [256KB, ...)    last-index arrays (int), 2*N_USERS + N_CAS
//   [WS_WID_OFF)    per-edge winner-id arrays (int), 3 x B
//   [WS_TRASH_OFF)  per-block trash rows, 2048 x 128 f32 (non-winner sink)
static constexpr size_t WP_HALFS     = (size_t)3 * 8 * 8 * 512;  // 98,304 halfs = 192 KB
static constexpr size_t WS_LAST_OFF  = 256 * 1024;
static constexpr size_t WS_WID_OFF   = WS_LAST_OFF + ((size_t)N_USERS * 2 + N_CAS) * 4;
static constexpr size_t WS_TRASH_OFF = WS_WID_OFF + (size_t)3 * B_EDGES * 4;

__global__ void zero_f32_kernel(float* __restrict__ p, size_t n) {
    size_t i = (size_t)blockIdx.x * blockDim.x + threadIdx.x;
    size_t stride = (size_t)gridDim.x * blockDim.x;
    for (; i < n; i += stride) p[i] = 0.0f;
}

__global__ void init_last_kernel(int* __restrict__ p, size_t n) {
    size_t i = (size_t)blockIdx.x * blockDim.x + threadIdx.x;
    size_t stride = (size_t)gridDim.x * blockDim.x;
    for (; i < n; i += stride) p[i] = -1;
}

__global__ void last_argmax_kernel(const int* __restrict__ src,
                                   const int* __restrict__ dst,
                                   const int* __restrict__ cas,
                                   int* __restrict__ lastS,
                                   int* __restrict__ lastD,
                                   int* __restrict__ lastC,
                                   int n) {
    int i = blockIdx.x * blockDim.x + threadIdx.x;
    if (i >= n) return;
    atomicMax(&lastS[src[i]], i);
    atomicMax(&lastD[dst[i]], i);
    atomicMax(&lastC[cas[i]], i);
}

// Per-edge winner ids: wid[i] = (this edge is the last one touching its node) ? node : -1
__global__ void win_ids_kernel(const int* __restrict__ src,
                               const int* __restrict__ dst,
                               const int* __restrict__ cas,
                               const int* __restrict__ lastS,
                               const int* __restrict__ lastD,
                               const int* __restrict__ lastC,
                               int* __restrict__ wS,
                               int* __restrict__ wD,
                               int* __restrict__ wC,
                               int n) {
    int i = blockIdx.x * blockDim.x + threadIdx.x;
    if (i >= n) return;
    int s = src[i]; wS[i] = (lastS[s] == i) ? s : -1;
    int d = dst[i]; wD[i] = (lastD[d] == i) ? d : -1;
    int c = cas[i]; wC[i] = (lastC[c] == i) ? c : -1;
}

__global__ void scatter_time_mask_kernel(const int* __restrict__ wS,
                                         const int* __restrict__ wD,
                                         const int* __restrict__ wC,
                                         const float* __restrict__ et,
                                         float* __restrict__ out,
                                         int n) {
    int i = blockIdx.x * blockDim.x + threadIdx.x;
    if (i >= n) return;
    int s = wS[i];
    if (s >= 0) { out[OFF_SRC_T + s] = et[i]; out[OFF_SRC_MK + s] = 1.0f; }
    int d = wD[i];
    if (d >= 0) { out[OFF_DST_T + d] = et[i]; out[OFF_DST_MK + d] = 1.0f; }
    int c = wC[i];
    if (c >= 0) { out[OFF_CAS_T + c] = et[i]; out[OFF_CAS_MK + c] = 1.0f; }
}

// Pack f32 weights into f16 B-fragment lane layout:
//   wp[((type*8 + kk)*8 + ct)*512 + lane*16 + j]
__global__ void pack_weights_kernel(const float* __restrict__ Ws,
                                    const float* __restrict__ Wd,
                                    const float* __restrict__ Wc,
                                    _Float16* __restrict__ wp) {
    int idx = blockIdx.x * blockDim.x + threadIdx.x;
    if (idx >= (int)WP_HALFS) return;
    int j    = idx & 15;
    int lane = (idx >> 4) & 31;
    int ct   = (idx >> 9) & 7;
    int kk   = (idx >> 12) & 7;
    int type = idx >> 15;
    int kbase = kk * 32 + ((lane & 16) ? 8 : 0);
    int k = kbase + ((j < 8) ? j : (8 + j));   // second half: kbase+16+(j-8)
    int n = ct * 16 + (lane & 15);
    const float* W = (type == 0) ? Ws : ((type == 1) ? Wd : Wc);
    wp[idx] = (_Float16)W[k * D_MSG + n];
}

// One message type: 4 column tiles x 8 k-steps = 32 WMMAs. Shared raw A
// fragments (kk 0..5) come from the caller; only the type-specific time
// fragments (kk 6,7) are loaded here. T is compile-time so weight offsets
// fold. Stores are branch-free: non-winner rows go to a per-block trash row.
template<int T>
__device__ __forceinline__ void gemm_type(const v16h* __restrict__ wpv,
                                          const v16h (&aRaw)[6],
                                          const _Float16* __restrict__ sTrow,
                                          const int* __restrict__ wid_arr,
                                          const float* __restrict__ biasp,
                                          float* __restrict__ o,
                                          float* __restrict__ trash,
                                          int lane, int lan16, int hi,
                                          int chalf, int i0) {
    // Type-specific time-encoding A fragments (K = 192..255)
    v16h aT[2];
    #pragma unroll
    for (int t = 0; t < 2; ++t) {
        const _Float16* p = sTrow + t * 32 + hi;
        #pragma unroll
        for (int j = 0; j < 8; ++j) { aT[t][j] = p[j]; aT[t][8 + j] = p[16 + j]; }
    }
    // Winner ids for the 8 rows this lane holds (contiguous, 32B-aligned).
    // Pre-select destination row base pointers once: winner -> out row,
    // non-winner -> trash row. Store loop below is fully unconditional.
    float* rowp[8];
    {
        const int4* w4 = (const int4*)(wid_arr + i0 + hi);
        int4 w0 = w4[0], w1 = w4[1];
        int wid[8] = { w0.x, w0.y, w0.z, w0.w, w1.x, w1.y, w1.z, w1.w };
        #pragma unroll
        for (int r = 0; r < 8; ++r) {
            rowp[r] = (wid[r] >= 0) ? (o + (size_t)wid[r] * D_MSG) : trash;
        }
    }
    #pragma unroll
    for (int c4 = 0; c4 < 4; ++c4) {
        const int ct = chalf * 4 + c4;
        const int n  = ct * 16 + lan16;

        v8f acc;
        #pragma unroll
        for (int r = 0; r < 8; ++r) acc[r] = 0.0f;

        #pragma unroll
        for (int kk = 0; kk < 6; ++kk) {
            const v16h b = wpv[(size_t)((T * 8 + kk) * 8 + ct) * 32 + lane];
            acc = __builtin_amdgcn_wmma_f32_16x16x32_f16(
                    false, aRaw[kk], false, b, (short)0, acc, false, false);
        }
        #pragma unroll
        for (int kk = 6; kk < 8; ++kk) {
            const v16h b = wpv[(size_t)((T * 8 + kk) * 8 + ct) * 32 + lane];
            acc = __builtin_amdgcn_wmma_f32_16x16x32_f16(
                    false, aT[kk - 6], false, b, (short)0, acc, false, false);
        }

        const float bias = biasp[n];
        #pragma unroll
        for (int r = 0; r < 8; ++r) {
            rowp[r][n] = acc[r] + bias;
        }
    }
}

// Fused gather + time-encode + 3x GEMM [64,256]x[256,128] via WMMA f16->f32.
// Block: 256 threads (8 waves), 64 edges. Each wave owns (rowTile = wave/2,
// ctHalf = wave&1) and runs all 3 message types -> 96 WMMAs per wave.
__global__ void __launch_bounds__(256)
fused_msg_wmma_kernel(const int*   __restrict__ src,
                      const int*   __restrict__ dst,
                      const int*   __restrict__ cas,
                      const float* __restrict__ edge_times,
                      const float* __restrict__ pub_times,
                      const float* __restrict__ u_src,
                      const float* __restrict__ u_dst,
                      const float* __restrict__ c_state,
                      const float* __restrict__ u_last_upd,
                      const float* __restrict__ w_tu,
                      const float* __restrict__ b_tu,
                      const float* __restrict__ w_tc,
                      const float* __restrict__ b_tc,
                      const float* __restrict__ b_src,
                      const float* __restrict__ b_dst,
                      const float* __restrict__ b_cas,
                      const _Float16* __restrict__ wp,
                      const int*   __restrict__ wS,
                      const int*   __restrict__ wD,
                      const int*   __restrict__ wC,
                      float* __restrict__ trash_base,
                      float* __restrict__ out) {
    __shared__ _Float16 sRaw[64][208];       // cols 0..191 (stride 416B = 26*16)
    __shared__ _Float16 sTime[3][64][72];    // cols 0..63  (stride 144B = 9*16)

    const int r0 = blockIdx.x * 64;

    // ---- Stage 1: gather raw states -> LDS (f16) ----
    for (int e = threadIdx.x; e < 64 * 192; e += 256) {
        int row = e / 192;
        int col = e - row * 192;
        int gr  = r0 + row;
        float v;
        if (col < 64)        v = u_src[(size_t)src[gr] * 64 + col];
        else if (col < 128)  v = u_dst[(size_t)dst[gr] * 64 + (col - 64)];
        else                 v = c_state[(size_t)cas[gr] * 64 + (col - 128)];
        sRaw[row][col] = (_Float16)v;
    }
    // ---- Stage 1b: time encodings cos(dt*w+b) -> LDS (f16) ----
    for (int e = threadIdx.x; e < 3 * 64 * 64; e += 256) {
        int type = e >> 12;
        int rem  = e & 4095;
        int row  = rem >> 6;
        int col  = rem & 63;
        int gr   = r0 + row;
        float et = edge_times[gr];
        float dt;
        if (type == 0)      dt = et - u_last_upd[src[gr]];
        else if (type == 1) dt = et - u_last_upd[dst[gr]];
        else                dt = et - pub_times[gr];
        const float* w = (type < 2) ? w_tu : w_tc;
        const float* b = (type < 2) ? b_tu : b_tc;
        sTime[type][row][col] = (_Float16)cosf(dt * w[col] + b[col]);
    }
    __syncthreads();

    const int wave  = threadIdx.x >> 5;
    const int lane  = threadIdx.x & 31;
    const int lan16 = lane & 15;
    const int hi    = (lane & 16) ? 8 : 0;

    const int rt    = wave >> 1;      // row tile 0..3
    const int chalf = wave & 1;       // column-tile half 0..1
    const int arow  = rt * 16 + lan16;
    const int i0    = r0 + rt * 16;

    float* trash = trash_base + (size_t)blockIdx.x * D_MSG;

    // Shared raw A fragments (K = 0..191), identical for all 3 types.
    v16h aRaw[6];
    #pragma unroll
    for (int kk = 0; kk < 6; ++kk) {
        const _Float16* p = &sRaw[arow][kk * 32 + hi];
        #pragma unroll
        for (int j = 0; j < 8; ++j) { aRaw[kk][j] = p[j]; aRaw[kk][8 + j] = p[16 + j]; }
    }

    const v16h* wpv = (const v16h*)wp;

    gemm_type<0>(wpv, aRaw, &sTime[0][arow][0], wS, b_src, out + OFF_SRC_M, trash,
                 lane, lan16, hi, chalf, i0);
    gemm_type<1>(wpv, aRaw, &sTime[1][arow][0], wD, b_dst, out + OFF_DST_M, trash,
                 lane, lan16, hi, chalf, i0);
    gemm_type<2>(wpv, aRaw, &sTime[2][arow][0], wC, b_cas, out + OFF_CAS_M, trash,
                 lane, lan16, hi, chalf, i0);
}

extern "C" void kernel_launch(void* const* d_in, const int* in_sizes, int n_in,
                              void* d_out, int out_size, void* d_ws, size_t ws_size,
                              hipStream_t stream) {
    const int*   src        = (const int*)  d_in[0];
    const int*   dst        = (const int*)  d_in[1];
    const int*   cas        = (const int*)  d_in[2];
    const float* edge_times = (const float*)d_in[3];
    const float* pub_times  = (const float*)d_in[4];
    const float* u_src      = (const float*)d_in[5];
    const float* u_dst      = (const float*)d_in[6];
    const float* c_state    = (const float*)d_in[7];
    const float* u_last     = (const float*)d_in[8];
    const float* w_tu       = (const float*)d_in[9];
    const float* b_tu       = (const float*)d_in[10];
    const float* w_tc       = (const float*)d_in[11];
    const float* b_tc       = (const float*)d_in[12];
    const float* W_src      = (const float*)d_in[13];
    const float* b_src      = (const float*)d_in[14];
    const float* W_dst      = (const float*)d_in[15];
    const float* b_dst      = (const float*)d_in[16];
    const float* W_cas      = (const float*)d_in[17];
    const float* b_cas      = (const float*)d_in[18];

    float* out = (float*)d_out;

    _Float16* wp = (_Float16*)d_ws;
    int* lastS = (int*)((char*)d_ws + WS_LAST_OFF);
    int* lastD = lastS + N_USERS;
    int* lastC = lastD + N_USERS;
    int* wS    = (int*)((char*)d_ws + WS_WID_OFF);
    int* wD    = wS + B_EDGES;
    int* wC    = wD + B_EDGES;
    float* trash_base = (float*)((char*)d_ws + WS_TRASH_OFF);
    const size_t n_last = (size_t)N_USERS * 2 + N_CAS;

    // 1) zero the (mostly-empty) outputs
    {
        size_t n = (size_t)out_size;
        int blocks = (int)((n + 256 * 8 - 1) / (256 * 8));
        zero_f32_kernel<<<blocks, 256, 0, stream>>>(out, n);
    }
    // 2) init last-index arrays to -1
    init_last_kernel<<<(int)((n_last + 255) / 256), 256, 0, stream>>>(lastS, n_last);
    // 3) pack weights into f16 fragment layout
    pack_weights_kernel<<<(int)((WP_HALFS + 255) / 256), 256, 0, stream>>>(
        W_src, W_dst, W_cas, wp);
    // 4) per-node argmax of batch position
    last_argmax_kernel<<<(B_EDGES + 255) / 256, 256, 0, stream>>>(
        src, dst, cas, lastS, lastD, lastC, B_EDGES);
    // 5) per-edge winner ids
    win_ids_kernel<<<(B_EDGES + 255) / 256, 256, 0, stream>>>(
        src, dst, cas, lastS, lastD, lastC, wS, wD, wC, B_EDGES);
    // 6) fused gather + time-encode + 3x WMMA GEMM + winner scatter
    fused_msg_wmma_kernel<<<B_EDGES / 64, 256, 0, stream>>>(
        src, dst, cas, edge_times, pub_times,
        u_src, u_dst, c_state, u_last,
        w_tu, b_tu, w_tc, b_tc,
        b_src, b_dst, b_cas,
        wp, wS, wD, wC, trash_base, out);
    // 7) winner times + masks
    scatter_time_mask_kernel<<<(B_EDGES + 255) / 256, 256, 0, stream>>>(
        wS, wD, wC, edge_times, out, B_EDGES);
}
